// descriptor_loss_37477884625612
// MI455X (gfx1250) — matrix operand order repaired and checked by reading, not compiled, not run on hardware
//
#include <hip/hip_runtime.h>

#define USE_ASYNC_LDS 1

typedef __attribute__((ext_vector_type(16))) _Float16 v16h;
typedef __attribute__((ext_vector_type(8)))  float    v8f;

#define MARGIN   0.5f
#define MAX_DIST 10000.0f
#define BIG      1000000000.0f

constexpr int BATCH = 4;
constexpr int DDIM  = 256;
constexpr int NN    = 4096;
constexpr int NP1   = NN + 1;

// Order-preserving float <-> uint mapping so integer atomicMax/Min implement
// exact float max/min (handles negatives from f16 rounding of dist=2-2*dot).
__device__ __forceinline__ unsigned enc(float f) {
  unsigned u = __float_as_uint(f);
  return (u >> 31) ? ~u : (u | 0x80000000u);
}
__device__ __forceinline__ float dec(unsigned k) {
  unsigned u = (k >> 31) ? (k & 0x7FFFFFFFu) : ~k;
  return __uint_as_float(u);
}

// Pure-VALU 16-lane reductions via DPP16 (no LDS, no ds waits).
__device__ __forceinline__ float row16_max(float v) {
  int x;
  x = __builtin_amdgcn_update_dpp(0, __float_as_int(v), 0xB1, 0xf, 0xf, true);
  v = fmaxf(v, __int_as_float(x));
  x = __builtin_amdgcn_update_dpp(0, __float_as_int(v), 0x4E, 0xf, 0xf, true);
  v = fmaxf(v, __int_as_float(x));
  x = __builtin_amdgcn_update_dpp(0, __float_as_int(v), 0x141, 0xf, 0xf, true);
  v = fmaxf(v, __int_as_float(x));
  x = __builtin_amdgcn_update_dpp(0, __float_as_int(v), 0x140, 0xf, 0xf, true);
  v = fmaxf(v, __int_as_float(x));
  return v;
}
__device__ __forceinline__ float row16_min(float v) {
  int x;
  x = __builtin_amdgcn_update_dpp(0, __float_as_int(v), 0xB1, 0xf, 0xf, true);
  v = fminf(v, __int_as_float(x));
  x = __builtin_amdgcn_update_dpp(0, __float_as_int(v), 0x4E, 0xf, 0xf, true);
  v = fminf(v, __int_as_float(x));
  x = __builtin_amdgcn_update_dpp(0, __float_as_int(v), 0x141, 0xf, 0xf, true);
  v = fminf(v, __int_as_float(x));
  x = __builtin_amdgcn_update_dpp(0, __float_as_int(v), 0x140, 0xf, 0xf, true);
  v = fminf(v, __int_as_float(x));
  return v;
}

#if USE_ASYNC_LDS
// CDNA5 async global->LDS copy (VGLOBAL opcode 98), tracked by ASYNCcnt.
// %0 = VGPR with LDS byte offset, %1 = VGPR pair with 64-bit global address.
__device__ __forceinline__ void async_ld128(unsigned lds_off, const float* g) {
  asm volatile("global_load_async_to_lds_b128 %0, %1, off"
               :
               : "v"(lds_off), "v"(g)
               : "memory");
}
__device__ __forceinline__ void wait_async0() {
  asm volatile("s_wait_asynccnt 0x0" ::: "memory");
}
// Generic (flat) pointers to LDS carry the byte offset in addr[31:0]
// (ISA 10.2 aperture mapping), so truncation yields the DS address.
__device__ __forceinline__ unsigned lds_off(const void* p) {
  return (unsigned)(uintptr_t)p;
}
#endif

__global__ __launch_bounds__(256) void init_keys(unsigned* posk, unsigned* mnk) {
  int i = blockIdx.x * blockDim.x + threadIdx.x;
  int total = 2 * BATCH * NN;
  if (i < total) {
    posk[i] = enc(-BIG);
    mnk[i]  = enc(BIG);
  }
}

// Block: 256 threads = 8 waves. Block tile: 64 rows (n) x 32 cols (m).
// Wave w computes the 16x16 tile at (tw = w>>1, tc = w&1).
// Double-buffered LDS staging (async global->LDS) overlaps chunk c+1 loads
// with chunk c WMMAs; one barrier per chunk.
template <int PASS>
__global__ __launch_bounds__(256) void dist_pass(
    const float* __restrict__ desc0, const float* __restrict__ desc1,
    const float* __restrict__ mat, unsigned* __restrict__ pos0k,
    unsigned* __restrict__ pos1k, unsigned* __restrict__ mn0k,
    unsigned* __restrict__ mn1k) {
  __shared__ float lA[2][32 * 64];  // [buf][k][n_local]
  __shared__ float lB[2][32 * 32];  // [buf][k][m_local]

  const int b  = blockIdx.z;
  const int n0 = blockIdx.x * 64;
  const int m0 = blockIdx.y * 32;

  const int tid  = threadIdx.x;
  const int wave = tid >> 5;
  const int lane = tid & 31;
  const int half = lane >> 4;  // 0 or 1
  const int l16  = lane & 15;
  const int tw   = wave >> 1;  // 0..3 (n sub-tile)
  const int tc   = wave & 1;   // 0..1 (m sub-tile)

  const float* A  = desc0 + (size_t)b * DDIM * NN;
  const float* Bm = desc1 + (size_t)b * DDIM * NN;

  // Per-thread float4 slices of the 32x64 A chunk and 32x32 B chunk.
  const int ka0 = tid >> 4, na4 = (tid & 15) * 4;  // A: k = ka0, ka0+16
  const int kb0 = tid >> 3, mb4 = (tid & 7) * 4;   // B: k = kb0

  auto stage = [&](int kk, int buf) {
    const float* ga0 = &A[(size_t)(kk + ka0) * NN + n0 + na4];
    const float* ga1 = &A[(size_t)(kk + ka0 + 16) * NN + n0 + na4];
    const float* gb0 = &Bm[(size_t)(kk + kb0) * NN + m0 + mb4];
#if USE_ASYNC_LDS
    async_ld128(lds_off(&lA[buf][ka0 * 64 + na4]), ga0);
    async_ld128(lds_off(&lA[buf][(ka0 + 16) * 64 + na4]), ga1);
    async_ld128(lds_off(&lB[buf][kb0 * 32 + mb4]), gb0);
#else
    float4 a0 = *(const float4*)ga0;
    float4 a1 = *(const float4*)ga1;
    float4 b0 = *(const float4*)gb0;
    *(float4*)&lA[buf][ka0 * 64 + na4]        = a0;
    *(float4*)&lA[buf][(ka0 + 16) * 64 + na4] = a1;
    *(float4*)&lB[buf][kb0 * 32 + mb4]        = b0;
#endif
  };

  stage(0, 0);

  // Prefetch the epilogue operands while the first stage is in flight.
  // ov is a 268MB single-use stream: non-temporal so it doesn't evict the
  // heavily reused desc tiles from L2.
  const float* ovb = mat + (size_t)b * NP1 * NP1;
  const int mcol   = m0 + tc * 16 + l16;
  float ovv[8];
#pragma unroll
  for (int r = 0; r < 8; ++r) {
    int n  = n0 + tw * 16 + 8 * half + r;
    ovv[r] = __builtin_nontemporal_load(&ovb[(size_t)n * NP1 + mcol]);
  }
  float p0v[8];
  float p1 = 0.0f;
  if (PASS == 1) {
#pragma unroll
    for (int r = 0; r < 8; ++r) {
      int n  = n0 + tw * 16 + 8 * half + r;
      p0v[r] = dec(pos0k[b * NN + n]);
    }
    p1 = dec(pos1k[b * NN + mcol]);
  }

#if USE_ASYNC_LDS
  wait_async0();
#endif
  __syncthreads();

  v8f c = {};
#pragma unroll
  for (int ch = 0; ch < 8; ++ch) {
    const int cur = ch & 1;
    if (ch + 1 < 8) stage((ch + 1) * 32, cur ^ 1);

    // Pack per-lane WMMA fragments per CDNA5 16-bit layouts:
    // A (16x32): lane<16 -> M=l16, K in {0..7,16..23}; lane>=16 -> K+8.
    // B (32x16): N=l16, lane<16 -> K in {0..7,16..23}; lane>=16 -> K+8.
    v16h af, bf;
#pragma unroll
    for (int p = 0; p < 16; ++p) {
      int ka = ((p < 8) ? p : p + 8) + 8 * half;
      af[p]  = (_Float16)lA[cur][ka * 64 + tw * 16 + l16];
      int kb = ((p < 8) ? p : p + 8) + 8 * half;
      bf[p]  = (_Float16)lB[cur][kb * 32 + tc * 16 + l16];
    }
    c = __builtin_amdgcn_wmma_f32_16x16x32_f16(false, af, false, bf, (short)0,
                                               c, false, false);
#if USE_ASYNC_LDS
    wait_async0();
#endif
    __syncthreads();
  }

  // C layout: c[r] -> row M = r + 8*half, col N = l16 (within 16x16 tile).
  if (PASS == 0) {
    float colmax = -BIG;
#pragma unroll
    for (int r = 0; r < 8; ++r) {
      float dist = 2.0f - 2.0f * c[r];
      float dp   = (ovv[r] > 0.3f) ? dist : 0.0f;
      float rm   = row16_max(dp);
      if (l16 == 0)
        atomicMax(&pos0k[b * NN + n0 + tw * 16 + 8 * half + r], enc(rm));
      colmax = fmaxf(colmax, dp);
    }
    // Both halves contribute their own partial column max (no cross-half op).
    atomicMax(&pos1k[b * NN + mcol], enc(colmax));
  } else {
    float colmin = BIG;
#pragma unroll
    for (int r = 0; r < 8; ++r) {
      float dist = 2.0f - 2.0f * c[r];
      float neg  = (ovv[r] <= 0.0f) ? dist : MAX_DIST;
      float p0   = p0v[r];
      float cand = ((neg > p0) && (neg < p0 + MARGIN)) ? neg : BIG;
      float rm   = row16_min(cand);
      if (l16 == 0)
        atomicMin(&mn0k[b * NN + n0 + tw * 16 + 8 * half + r], enc(rm));
      float cand1 = ((neg > p1) && (neg < p1 + MARGIN)) ? neg : BIG;
      colmin      = fminf(colmin, cand1);
    }
    atomicMin(&mn1k[b * NN + mcol], enc(colmin));
  }
}

__global__ __launch_bounds__(256) void final_reduce(
    const unsigned* __restrict__ posk, const unsigned* __restrict__ mnk,
    float* __restrict__ out) {
  __shared__ float smx[256], smn[256], ssum[256], scnt[256];
  float mx = -BIG, mnv = BIG, sum = 0.0f, cnt = 0.0f;
  for (int i = threadIdx.x; i < 2 * BATCH * NN; i += 256) {
    float pos = dec(posk[i]);
    float mn  = dec(mnk[i]);
    // has <=> mn < BIG (any accepted neg is < pos+0.5 <= ~4.5)
    if ((pos > 0.0f) && (mn < BIG)) {
      mx  = fmaxf(mx, pos);
      mnv = fminf(mnv, mn);
      float l = pos - mn + 1.0f;
      sum += (l > 0.0f) ? l : 0.0f;
      cnt += 1.0f;
    }
  }
  int t   = threadIdx.x;
  smx[t]  = mx;
  smn[t]  = mnv;
  ssum[t] = sum;
  scnt[t] = cnt;
  __syncthreads();
  for (int s = 128; s > 0; s >>= 1) {
    if (t < s) {
      smx[t]  = fmaxf(smx[t], smx[t + s]);
      smn[t]  = fminf(smn[t], smn[t + s]);
      ssum[t] += ssum[t + s];
      scnt[t] += scnt[t + s];
    }
    __syncthreads();
  }
  if (t == 0) {
    out[0] = ssum[0] / fmaxf(scnt[0], 1.0f);
    out[1] = smx[0];
    out[2] = smn[0];
  }
}

extern "C" void kernel_launch(void* const* d_in, const int* in_sizes, int n_in,
                              void* d_out, int out_size, void* d_ws,
                              size_t ws_size, hipStream_t stream) {
  const float* desc0 = (const float*)d_in[0];
  const float* desc1 = (const float*)d_in[1];
  const float* mat   = (const float*)d_in[2];
  float* out         = (float*)d_out;

  unsigned* posk  = (unsigned*)d_ws;           // 2*B*N keys (pos0 | pos1)
  unsigned* mnk   = posk + 2 * BATCH * NN;     // 2*B*N keys (mn0 | mn1)
  unsigned* pos0k = posk;
  unsigned* pos1k = posk + BATCH * NN;
  unsigned* mn0k  = mnk;
  unsigned* mn1k  = mnk + BATCH * NN;

  int total = 2 * BATCH * NN;
  init_keys<<<(total + 255) / 256, 256, 0, stream>>>(posk, mnk);

  dim3 grid(NN / 64, NN / 32, BATCH);
  dist_pass<0><<<grid, 256, 0, stream>>>(desc0, desc1, mat, pos0k, pos1k, mn0k,
                                         mn1k);
  dist_pass<1><<<grid, 256, 0, stream>>>(desc0, desc1, mat, pos0k, pos1k, mn0k,
                                         mn1k);
  final_reduce<<<1, 256, 0, stream>>>(posk, mnk, out);
}
